// ClapAudioSelfAttention_55336358643216
// MI455X (gfx1250) — compile-verified, matrix operand order checked by reading, not run on hardware
//
#include <hip/hip_runtime.h>
#include <hip/hip_bf16.h>

typedef __attribute__((ext_vector_type(16))) _Float16 v16h;
typedef __attribute__((ext_vector_type(8)))  _Float16 v8h;
typedef __attribute__((ext_vector_type(4)))  _Float16 v4h;
typedef __attribute__((ext_vector_type(8)))  float    v8f;

#define NWIN   2048
#define NTOK   64
#define CDIM   256
#define NHEAD  8
#define DHEAD  32
#define SCALE  0.17677669529663687f   // 1/sqrt(32)

// ---- LDS layout (units: _Float16 elements) ----
// Region 0 (aliased): X staging [64][272] (17408) then P scratch 8*[64][72] (36864)
// Qs: 8 * [64][40]  = 20480
// Ks: 8 * [64][40]  = 20480
// Vt: 8 * [32][72]  = 18432   (V stored transposed: [chan][tok])
#define XS_STRIDE 272
#define QK_STRIDE 40
#define VT_STRIDE 72
#define P_STRIDE  72
#define PS_OFF    0
#define QS_OFF    36864
#define KS_OFF    (QS_OFF + 20480)
#define VT_OFF    (KS_OFF + 20480)
#define LDS_HALVES (VT_OFF + 18432)          // 96256 halves
#define LDS_BYTES  (LDS_HALVES * 2)          // 192512 bytes < 320KB/WGP

__device__ __forceinline__ v8f vzero() {
  v8f z;
#pragma unroll
  for (int i = 0; i < 8; ++i) z[i] = 0.0f;
  return z;
}

__device__ __forceinline__ v8f wmma16(v16h a, v16h b, v8f c) {
  // D = A(16x32 f16) * B(32x16 f16) + C(16x16 f32)
  return __builtin_amdgcn_wmma_f32_16x16x32_f16(
      /*neg_a=*/false, a, /*neg_b=*/false, b,
      /*c_mod=*/(short)0, c, /*reuse_a=*/false, /*reuse_b=*/false);
}

// A/B fragment load for v_wmma_*_16x16x32 (16-bit):
//   lanes 0-15  hold K = kbase+{0..7} and kbase+{16..23}  (kbase = 0)
//   lanes 16-31 hold K = kbase+{0..7} and kbase+{16..23}  (kbase = 8)
// base points at the start of this lane's matrix row (row-major, contiguous K).
__device__ __forceinline__ v16h load_frag(const _Float16* base, int kbase) {
  v8h lo = *(const v8h*)(base + kbase);
  v8h hi = *(const v8h*)(base + kbase + 16);
  v16h r;
#pragma unroll
  for (int i = 0; i < 8; ++i) { r[i] = lo[i]; r[i + 8] = hi[i]; }
  return r;
}

// ---------------------------------------------------------------------------
// Kernel 0a: weights f32 -> f16 (row-major preserved; rows of W are exactly
// the B-fragment lane rows for out = X @ W^T).
// ---------------------------------------------------------------------------
__global__ void cvt_weights(const float* __restrict__ wq,
                            const float* __restrict__ wk,
                            const float* __restrict__ wv,
                            _Float16* __restrict__ q16,
                            _Float16* __restrict__ k16,
                            _Float16* __restrict__ v16) {
  int i = blockIdx.x * 256 + threadIdx.x;   // 0 .. 65535
  q16[i] = (_Float16)wq[i];
  k16[i] = (_Float16)wk[i];
  v16[i] = (_Float16)wv[i];
}

// ---------------------------------------------------------------------------
// Kernel 0b: combined additive bias  cb[w][h][q][k] =
//            mask[w][q][k] + bias_table[rel_index[q][k]][h]   (f32, 8 MB)
// ---------------------------------------------------------------------------
__global__ void build_cb(const float* __restrict__ mask,
                         const float* __restrict__ btab,
                         const int*   __restrict__ ridx,
                         float* __restrict__ cb) {
  int i = blockIdx.x * 256 + threadIdx.x;   // 0 .. 2097151
  int k = i & 63;
  int q = (i >> 6) & 63;
  int h = (i >> 12) & 7;
  int w = i >> 15;
  cb[i] = mask[(w << 12) + (q << 6) + k] + btab[ridx[(q << 6) + k] * NHEAD + h];
}

// ---------------------------------------------------------------------------
// Main fused kernel: 1 block = 1 window (8 waves, wave h = head h)
// ---------------------------------------------------------------------------
__global__ __launch_bounds__(256, 1)
void swin_window_attn(const float* __restrict__ X,
                      const float* __restrict__ bq,
                      const float* __restrict__ bk,
                      const float* __restrict__ bv,
                      const _Float16* __restrict__ Wq16,
                      const _Float16* __restrict__ Wk16,
                      const _Float16* __restrict__ Wv16,
                      const float* __restrict__ cb,
                      float* __restrict__ out) {
  extern __shared__ _Float16 lds[];

  const int b     = blockIdx.x;
  const int tid   = threadIdx.x;
  const int head  = tid >> 5;          // wave id == head id (wave32)
  const int lane  = tid & 31;
  const int l15   = lane & 15;
  const int hihalf = lane >> 4;        // 0: lanes 0-15, 1: lanes 16-31
  const int kbase = hihalf * 8;        // fragment K sub-chunk base

  // ---------------- Phase 0: stage X (64x256 f32 -> f16 in LDS) ------------
  {
    const float4* src = (const float4*)(X + (size_t)b * NTOK * CDIM);
#pragma unroll
    for (int j = 0; j < 16; ++j) {
      int idx4 = j * 256 + tid;                  // coalesced across threads
      float4 v = src[idx4];
      int tok  = idx4 >> 6;                      // /64 float4 per row
      int chan = (idx4 & 63) << 2;
      v4h h4;
      h4[0] = (_Float16)v.x; h4[1] = (_Float16)v.y;
      h4[2] = (_Float16)v.z; h4[3] = (_Float16)v.w;
      *(v4h*)(lds + tok * XS_STRIDE + chan) = h4;
    }
  }
  __syncthreads();

  // ---------------- Phase 1: per-head QKV projection (WMMA) ----------------
  _Float16* Qs = lds + QS_OFF + head * (NTOK * QK_STRIDE);
  _Float16* Ks = lds + KS_OFF + head * (NTOK * QK_STRIDE);
  _Float16* Vt = lds + VT_OFF + head * (DHEAD * VT_STRIDE);
  const int hbase = head * DHEAD;                // output-channel base

  const _Float16* Ws[3] = {Wq16, Wk16, Wv16};
  const float*    bs[3] = {bq, bk, bv};

#pragma unroll
  for (int p = 0; p < 3; ++p) {
    const _Float16* W = Ws[p];
    v8f acc[4][2];
#pragma unroll
    for (int tm = 0; tm < 4; ++tm)
#pragma unroll
      for (int tn = 0; tn < 2; ++tn) acc[tm][tn] = vzero();

    for (int kc = 0; kc < CDIM; kc += 32) {      // K loop (8 iters)
      v16h bfr[2];
#pragma unroll
      for (int tn = 0; tn < 2; ++tn) {
        int co = hbase + tn * 16 + l15;          // W row = output channel
        bfr[tn] = load_frag(W + co * CDIM + kc, kbase);
      }
#pragma unroll
      for (int tm = 0; tm < 4; ++tm) {
        int tok = tm * 16 + l15;
        v16h afr = load_frag(lds + tok * XS_STRIDE + kc, kbase);
        acc[tm][0] = wmma16(afr, bfr[0], acc[tm][0]);
        acc[tm][1] = wmma16(afr, bfr[1], acc[tm][1]);
      }
    }

    // epilogue: +bias, (Q: *scale), f16 -> LDS (V transposed)
#pragma unroll
    for (int tn = 0; tn < 2; ++tn) {
      float bias = bs[p][hbase + tn * 16 + l15];
#pragma unroll
      for (int tm = 0; tm < 4; ++tm) {
#pragma unroll
        for (int r = 0; r < 8; ++r) {
          int tok  = tm * 16 + r + 8 * hihalf;   // C-layout row
          int chan = tn * 16 + l15;              // C-layout col (0..31)
          float val = acc[tm][tn][r] + bias;
          if (p == 0)      Qs[tok * QK_STRIDE + chan] = (_Float16)(val * SCALE);
          else if (p == 1) Ks[tok * QK_STRIDE + chan] = (_Float16)val;
          else             Vt[chan * VT_STRIDE + tok] = (_Float16)val;
        }
      }
    }
  }
  __syncthreads();   // all waves done reading X before P-scratch aliases it

  // ---------------- Phase 2: scores = Q K^T (+ bias), softmax --------------
  v16h kf[4];
#pragma unroll
  for (int tn = 0; tn < 4; ++tn)
    kf[tn] = load_frag(Ks + (tn * 16 + l15) * QK_STRIDE, kbase);

  v8f sc[4][4];
#pragma unroll
  for (int tm = 0; tm < 4; ++tm) {
    v16h qf = load_frag(Qs + (tm * 16 + l15) * QK_STRIDE, kbase);
#pragma unroll
    for (int tn = 0; tn < 4; ++tn)
      sc[tm][tn] = wmma16(qf, kf[tn], vzero());   // K=32 in one WMMA
  }

  // combined bias add (coalesced per half-wave: lane dim == k dim)
  const float* cbw = cb + (((size_t)(b & 63) * NHEAD + head) << 12);
#pragma unroll
  for (int tm = 0; tm < 4; ++tm)
#pragma unroll
    for (int r = 0; r < 8; ++r) {
      int q = tm * 16 + r + 8 * hihalf;
#pragma unroll
      for (int tn = 0; tn < 4; ++tn)
        sc[tm][tn][r] += cbw[(q << 6) + tn * 16 + l15];
    }

  // softmax over k (row lives in one 16-lane half across 4 tile columns)
#pragma unroll
  for (int tm = 0; tm < 4; ++tm) {
#pragma unroll
    for (int r = 0; r < 8; ++r) {
      float m = sc[tm][0][r];
      m = fmaxf(m, sc[tm][1][r]);
      m = fmaxf(m, sc[tm][2][r]);
      m = fmaxf(m, sc[tm][3][r]);
      m = fmaxf(m, __shfl_xor(m, 1, 16));
      m = fmaxf(m, __shfl_xor(m, 2, 16));
      m = fmaxf(m, __shfl_xor(m, 4, 16));
      m = fmaxf(m, __shfl_xor(m, 8, 16));
      float e0 = __expf(sc[tm][0][r] - m);
      float e1 = __expf(sc[tm][1][r] - m);
      float e2 = __expf(sc[tm][2][r] - m);
      float e3 = __expf(sc[tm][3][r] - m);
      float s = e0 + e1 + e2 + e3;
      s += __shfl_xor(s, 1, 16);
      s += __shfl_xor(s, 2, 16);
      s += __shfl_xor(s, 4, 16);
      s += __shfl_xor(s, 8, 16);
      float inv = 1.0f / s;
      sc[tm][0][r] = e0 * inv;
      sc[tm][1][r] = e1 * inv;
      sc[tm][2][r] = e2 * inv;
      sc[tm][3][r] = e3 * inv;
    }
  }

  // P (C-layout) -> LDS f16 (row-major) for A-fragment reload
  _Float16* Ps = lds + PS_OFF + head * (NTOK * P_STRIDE);
#pragma unroll
  for (int tm = 0; tm < 4; ++tm)
#pragma unroll
    for (int tn = 0; tn < 4; ++tn)
#pragma unroll
      for (int r = 0; r < 8; ++r) {
        int q = tm * 16 + r + 8 * hihalf;
        int k = tn * 16 + l15;
        Ps[q * P_STRIDE + k] = (_Float16)sc[tm][tn][r];
      }
  // wave-local LDS RAW: DS ops are in-order within a wave; compiler waits.

  // ---------------- Phase 3: ctx = P @ V  (K=64 -> 2 WMMA chain) -----------
  v16h vf[2][2];   // [chan tile][k chunk]
#pragma unroll
  for (int tn = 0; tn < 2; ++tn)
#pragma unroll
    for (int kc2 = 0; kc2 < 2; ++kc2)
      vf[tn][kc2] = load_frag(Vt + (tn * 16 + l15) * VT_STRIDE + kc2 * 32, kbase);

  v8f ctx[4][2];
#pragma unroll
  for (int tm = 0; tm < 4; ++tm)
#pragma unroll
    for (int tn = 0; tn < 2; ++tn) ctx[tm][tn] = vzero();

#pragma unroll
  for (int tm = 0; tm < 4; ++tm)
#pragma unroll
    for (int kc2 = 0; kc2 < 2; ++kc2) {
      v16h pf = load_frag(Ps + (tm * 16 + l15) * P_STRIDE + kc2 * 32, kbase);
      ctx[tm][0] = wmma16(pf, vf[0][kc2], ctx[tm][0]);
      ctx[tm][1] = wmma16(pf, vf[1][kc2], ctx[tm][1]);
    }

  // store ctx -> out[b][q][h*32+chan] (f32, coalesced per half-wave)
  float* outb = out + (size_t)b * NTOK * CDIM;
#pragma unroll
  for (int tm = 0; tm < 4; ++tm)
#pragma unroll
    for (int tn = 0; tn < 2; ++tn)
#pragma unroll
      for (int r = 0; r < 8; ++r) {
        int q    = tm * 16 + r + 8 * hihalf;
        int chan = hbase + tn * 16 + l15;
        outb[q * CDIM + chan] = ctx[tm][tn][r];
      }
}

// ---------------------------------------------------------------------------
extern "C" void kernel_launch(void* const* d_in, const int* in_sizes, int n_in,
                              void* d_out, int out_size, void* d_ws, size_t ws_size,
                              hipStream_t stream) {
  const float* X    = (const float*)d_in[0];
  const float* mask = (const float*)d_in[1];
  const float* Wq   = (const float*)d_in[2];
  const float* bq   = (const float*)d_in[3];
  const float* Wk   = (const float*)d_in[4];
  const float* bk   = (const float*)d_in[5];
  const float* Wv   = (const float*)d_in[6];
  const float* bv   = (const float*)d_in[7];
  const float* btab = (const float*)d_in[8];
  const int*   ridx = (const int*)d_in[9];

  // workspace: 3x f16 weights (384 KB) + combined bias f32 (8 MB)
  char* ws = (char*)d_ws;
  _Float16* Wq16 = (_Float16*)(ws + 0);
  _Float16* Wk16 = (_Float16*)(ws + 131072);
  _Float16* Wv16 = (_Float16*)(ws + 262144);
  float*    cbuf = (float*)   (ws + 393216);

  cvt_weights<<<256, 256, 0, stream>>>(Wq, Wk, Wv, Wq16, Wk16, Wv16);
  build_cb<<<8192, 256, 0, stream>>>(mask, btab, ridx, cbuf);
  swin_window_attn<<<NWIN, 256, LDS_BYTES, stream>>>(
      X, bq, bk, bv, Wq16, Wk16, Wv16, cbuf, (float*)d_out);
}
